// ModulatedConv2d_7378753814939
// MI455X (gfx1250) — compile-verified, matrix operand order checked by reading, not run on hardware
//
#include <hip/hip_runtime.h>
#include <hip/hip_bf16.h>
#include <stdint.h>

#define B_    16
#define CIN   512
#define COUT  512
#define SDIM  512
#define HH    64
#define WW    64
#define NPIX  (HH*WW)
#define CONV_SCALE 0.014731391274719739f   /* 1/sqrt(512*9) */
#define MOD_SCALE  0.044194173824159216f   /* 1/sqrt(512)   */
#define EPS_       1e-8f

typedef __attribute__((ext_vector_type(16))) __bf16 v16bf;
typedef __attribute__((ext_vector_type(8)))  __bf16 v8bf;
typedef __attribute__((ext_vector_type(8)))  float  v8f;
typedef __attribute__((ext_vector_type(4)))  int    v4i;

static __device__ __forceinline__ unsigned short f32_to_bf16(float f) {
    union { float f; unsigned u; } v; v.f = f;
    unsigned u = v.u;
    u = u + 0x7FFFu + ((u >> 16) & 1u);   // round-to-nearest-even
    return (unsigned short)(u >> 16);
}

static __device__ __forceinline__ v16bf cat16(v8bf lo, v8bf hi) {
    return __builtin_shufflevector(lo, hi, 0,1,2,3,4,5,6,7,8,9,10,11,12,13,14,15);
}

// ---------------------------------------------------------------------------
// s[b][c] = style[b] . (mod_weight[c] * MOD_SCALE) + mod_bias[c]
// ---------------------------------------------------------------------------
__global__ void mod_kernel(const float* __restrict__ style,
                           const float* __restrict__ mw,
                           const float* __restrict__ mb,
                           float* __restrict__ s) {
    int t = blockIdx.x * blockDim.x + threadIdx.x;
    if (t >= B_ * CIN) return;
    int b = t / CIN, c = t % CIN;
    const float* st = style + (size_t)b * SDIM;
    const float* w  = mw    + (size_t)c * SDIM;
    float acc = 0.f;
    for (int j = 0; j < SDIM; ++j) acc += st[j] * w[j];
    s[t] = acc * MOD_SCALE + mb[c];
}

// ---------------------------------------------------------------------------
// q2[co][ci] = sum_k weight[co][ci][k]^2 ; wbf[t][co][ci] = bf16(weight)
// ---------------------------------------------------------------------------
__global__ void wprep_kernel(const float* __restrict__ weight,
                             float* __restrict__ q2,
                             unsigned short* __restrict__ wbf) {
    int t = blockIdx.x * blockDim.x + threadIdx.x;
    if (t >= COUT * CIN) return;
    int co = t >> 9, ci = t & 511;
    const float* wp = weight + (size_t)(co * CIN + ci) * 9;
    float ss = 0.f;
#pragma unroll
    for (int k = 0; k < 9; ++k) {
        float v = wp[k];
        ss += v * v;
        wbf[(size_t)k * COUT * CIN + (size_t)co * CIN + ci] = f32_to_bf16(v);
    }
    q2[t] = ss;
}

// ---------------------------------------------------------------------------
// sigma[b][co] = rsqrt(CONV_SCALE^2 * sum_ci q2[co][ci]*s[b][ci]^2 + eps)
// ---------------------------------------------------------------------------
__global__ void demod_kernel(const float* __restrict__ q2,
                             const float* __restrict__ s,
                             float* __restrict__ sigma) {
    int t = blockIdx.x * blockDim.x + threadIdx.x;
    if (t >= B_ * COUT) return;
    int b = t / COUT, co = t % COUT;
    const float* q  = q2 + (size_t)co * CIN;
    const float* sb = s  + (size_t)b  * CIN;
    float acc = 0.f;
    for (int ci = 0; ci < CIN; ++ci) { float sv = sb[ci]; acc += q[ci] * sv * sv; }
    sigma[t] = rsqrtf(CONV_SCALE * CONV_SCALE * acc + EPS_);
}

// ---------------------------------------------------------------------------
// Implicit-GEMM modulated conv. Block: 256 threads (8 waves).
// Tile: 128 cout x 128 pixels (2 image rows). Weight tiles streamed into LDS
// with GLOBAL_LOAD_ASYNC_TO_LDS_B128 (ASYNCcnt), one 3-tap row group at a time.
// ---------------------------------------------------------------------------
__global__ __launch_bounds__(256, 1) void conv_kernel(
        const float* __restrict__ x,
        const unsigned short* __restrict__ wbf,
        const float* __restrict__ s,
        const float* __restrict__ sigma,
        float* __restrict__ out) {
    __shared__ __align__(16) unsigned short lds_x[4][66][32];   // [row][col][cin]
    __shared__ __align__(16) unsigned short lds_w[3][128][32];  // [tap][cout][cin]
    __shared__ float lds_s[CIN];
    __shared__ float lds_sig[128];

    const int tid  = threadIdx.x;
    const int lane = tid & 31;
    const int wave = tid >> 5;
    const int wm   = wave >> 1;      // 0..3 : 32-row M group
    const int wn   = wave & 1;       // 0..1 : 64-col N group
    const int half = lane >> 4;
    const int l16  = lane & 15;

    const int b   = blockIdx.z;
    const int co0 = blockIdx.y * 128;
    const int p0  = blockIdx.x * 128;
    const int h0  = p0 >> 6;         // first of the two output rows

    for (int i = tid; i < CIN; i += 256) lds_s[i] = s[(size_t)b * CIN + i];
    if (tid < 128) lds_sig[tid] = sigma[(size_t)b * COUT + co0 + tid];

    v8f acc[2][4] = {};

    for (int kc = 0; kc < CIN / 32; ++kc) {
        __syncthreads();   // previous x/w tiles fully consumed
        // ---- stage input tile (scaled by CONV_SCALE*s, f32->bf16) ----
        for (int i = tid; i < 4 * 66 * 32; i += 256) {
            int k   = i / (4 * 66);
            int rem = i - k * (4 * 66);
            int r   = rem / 66;
            int c   = rem - r * 66;
            int gr  = h0 - 1 + r;
            int gc  = c - 1;
            float v = 0.f;
            if ((unsigned)gr < 64u && (unsigned)gc < 64u) {
                size_t gidx = (((size_t)b * CIN + kc * 32 + k) * 64 + gr) * 64 + gc;
                v = x[gidx] * (CONV_SCALE * lds_s[kc * 32 + k]);
                if (kc + 1 < CIN / 32)
                    __builtin_prefetch(&x[gidx + (size_t)32 * 64 * 64], 0, 0);
            }
            lds_x[r][c][k] = f32_to_bf16(v);
        }
        for (int tr = 0; tr < 3; ++tr) {       // tap row group: taps tr*3 .. tr*3+2
            __syncthreads();   // (tr==0): x ds-stores visible; (tr>0): prior lds_w reads done
            // ---- async-copy 3 weight tiles (3 x 128 x 32 bf16 = 24 KB) ----
#pragma unroll
            for (int q = 0; q < 6; ++q) {
                int j   = q * 256 + tid;       // [0, 1536) b128 chunks
                int tq  = j >> 9;              // 0..2
                int rem = j & 511;
                int co  = rem >> 2, seg = rem & 3;
                const unsigned short* gsrc =
                    wbf + (size_t)(tr * 3 + tq) * COUT * CIN +
                    (size_t)(co0 + co) * CIN + kc * 32 + seg * 8;
                __builtin_amdgcn_global_load_async_to_lds_b128(
                    (v4i*)gsrc, (v4i*)&lds_w[tq][co][seg * 8], 0, 0);
            }
            asm volatile("s_wait_asynccnt 0x0" ::: "memory");  // this wave's copies landed
            __syncthreads();                                    // all waves' copies visible
#pragma unroll
            for (int tq = 0; tq < 3; ++tq) {   // dr = tr, dc = tq
                // ---- A fragments (16x32 bf16 each) ----
                const __bf16* row0 = (const __bf16*)&lds_w[tq][wm * 32 + l16][0];
                v16bf a0 = cat16(*(const v8bf*)(row0 + half * 8),
                                 *(const v8bf*)(row0 + 16 + half * 8));
                const __bf16* row1 = (const __bf16*)&lds_w[tq][wm * 32 + 16 + l16][0];
                v16bf a1 = cat16(*(const v8bf*)(row1 + half * 8),
                                 *(const v8bf*)(row1 + 16 + half * 8));
#pragma unroll
                for (int ni = 0; ni < 4; ++ni) {
                    int n  = wn * 64 + ni * 16 + l16;   // pixel within 128-tile
                    int r  = (n >> 6) + tr;             // 0..3
                    int cc = (n & 63) + tq;             // 0..65
                    const __bf16* p = (const __bf16*)&lds_x[r][cc][0];
                    v16bf bf = cat16(*(const v8bf*)(p + half * 8),
                                     *(const v8bf*)(p + 16 + half * 8));
                    acc[0][ni] = __builtin_amdgcn_wmma_f32_16x16x32_bf16(
                        false, a0, false, bf, (short)0, acc[0][ni], false, false);
                    acc[1][ni] = __builtin_amdgcn_wmma_f32_16x16x32_bf16(
                        false, a1, false, bf, (short)0, acc[1][ni], false, false);
                }
            }
        }
    }
    // ---- epilogue: scale by demod sigma, store f32 ----
#pragma unroll
    for (int mi = 0; mi < 2; ++mi) {
#pragma unroll
        for (int ni = 0; ni < 4; ++ni) {
            int n  = wn * 64 + ni * 16 + l16;
            int h  = h0 + (n >> 6);
            int wc = n & 63;
#pragma unroll
            for (int r = 0; r < 8; ++r) {
                int M = wm * 32 + mi * 16 + half * 8 + r;
                out[(((size_t)b * COUT + co0 + M) * 64 + h) * 64 + wc] =
                    acc[mi][ni][r] * lds_sig[M];
            }
        }
    }
}

// ---------------------------------------------------------------------------
extern "C" void kernel_launch(void* const* d_in, const int* in_sizes, int n_in,
                              void* d_out, int out_size, void* d_ws, size_t ws_size,
                              hipStream_t stream) {
    (void)in_sizes; (void)n_in; (void)out_size; (void)ws_size;
    const float* x      = (const float*)d_in[0];
    const float* style  = (const float*)d_in[1];
    const float* weight = (const float*)d_in[2];
    const float* mw     = (const float*)d_in[3];
    const float* mb     = (const float*)d_in[4];
    float* out = (float*)d_out;

    char* ws = (char*)d_ws;
    float* s     = (float*)(ws);                        // 16*512 f32   (32 KB)
    float* sigma = (float*)(ws + 32768);                // 16*512 f32   (32 KB)
    float* q2    = (float*)(ws + 65536);                // 512*512 f32  (1 MB)
    unsigned short* wbf = (unsigned short*)(ws + 65536 + 1048576); // 9*512*512 bf16 (4.7 MB)

    mod_kernel  <<<(B_ * CIN  + 255) / 256, 256, 0, stream>>>(style, mw, mb, s);
    wprep_kernel<<<(COUT * CIN + 255) / 256, 256, 0, stream>>>(weight, q2, wbf);
    demod_kernel<<<(B_ * COUT + 255) / 256, 256, 0, stream>>>(q2, s, sigma);

    dim3 grid(NPIX / 128, COUT / 128, B_);
    conv_kernel<<<grid, 256, 0, stream>>>(x, wbf, s, sigma, out);
}
